// NodeConv_19344532702267
// MI455X (gfx1250) — matrix operand order repaired and checked by reading, not compiled, not run on hardware
//
#include <hip/hip_runtime.h>
#include <hip/hip_bf16.h>
#include <stdint.h>

#define N_NODES 100000
#define N_EDGES 1600000
#define DIM     128
#define ROWS    32          // nodes per block in fused kernel
#define S0      264         // bf16 stride for 256-col act buffers (+8 pad -> 528B rows)
#define SW      136         // bf16 stride for 128-col transposed weights (+8 pad)
#define SH      132         // f32 stride for output staging (+4 pad)
#define EPS     1e-5f

// weight image: [128][S0] (W0^T) + 3x [128][SW] (W1^T,W2^T,Wo^T), bf16
#define WIMG_ELEMS (128 * S0 + 3 * 128 * SW)       // 86016 bf16
#define WIMG_BYTES (WIMG_ELEMS * 2)                // 172032 B
#define MSG_BYTES  ((size_t)N_NODES * DIM * 4)     // 51.2 MB

typedef __attribute__((ext_vector_type(16))) __bf16 v16bf;
typedef __attribute__((ext_vector_type(8)))  __bf16 v8bf;
typedef __attribute__((ext_vector_type(4)))  __bf16 v4bf;
typedef __attribute__((ext_vector_type(8)))  float  v8f;
typedef __attribute__((ext_vector_type(4)))  float  v4f;
typedef __attribute__((ext_vector_type(4)))  unsigned int u32x4;
typedef __attribute__((ext_vector_type(4)))  int  i32x4;
typedef __attribute__((ext_vector_type(8)))  int  i32x8;

#if defined(__has_builtin)
#if __has_builtin(__builtin_amdgcn_tensor_load_to_lds) && \
    __has_builtin(__builtin_amdgcn_s_wait_tensorcnt)
#define HAVE_TDM 1
#endif
#endif
#ifndef HAVE_TDM
#define HAVE_TDM 0
#endif

static __device__ inline __bf16 tobf(float f) {
    // round-to-nearest-even f32 -> bf16
    unsigned u = __builtin_bit_cast(unsigned, f);
    u += 0x7FFFu + ((u >> 16) & 1u);
    unsigned short h = (unsigned short)(u >> 16);
    return __builtin_bit_cast(__bf16, h);
}

// ---------------- kernel 1: zero the msg accumulator (in d_ws) ----------------
__global__ void zero_kernel(float* __restrict__ p, long n) {
    long i = ((long)blockIdx.x * blockDim.x + threadIdx.x) * 4;
    if (i < n) *(v4f*)(p + i) = (v4f){0.f, 0.f, 0.f, 0.f};
}

// ------- kernel 2: one-shot weight transpose + f32->bf16 into d_ws image ------
__global__ void prep_weights(const float* __restrict__ W0,
                             const float* __restrict__ Wh,
                             const float* __restrict__ Wo,
                             __bf16* __restrict__ wimg) {
    int idx = blockIdx.x * 256 + threadIdx.x;      // 81920 total
    if (idx >= 81920) return;
    if (idx < 256 * 128) {                         // W0: [256][128] -> [n][k] pad S0
        int k = idx >> 7, n = idx & 127;
        wimg[n * S0 + k] = tobf(W0[idx]);
    } else {
        int j = idx - 256 * 128;
        int mat = j >> 14;                         // 0: Wh[0], 1: Wh[1], 2: Wo
        int r = j & 16383;
        int k = r >> 7, n = r & 127;
        const float* src = (mat == 0) ? Wh : (mat == 1) ? (Wh + 16384) : Wo;
        __bf16* dst = wimg + 128 * S0 + mat * (128 * SW);
        dst[n * SW + k] = tobf(src[r]);
    }
}

// ---------------- kernel 3: scatter-sum edges onto destination nodes ----------
// one wave per edge, 4 channels per lane; NT loads (single-use stream)
__global__ void scatter_kernel(const float* __restrict__ e,
                               const long long* __restrict__ dst_idx,
                               float* __restrict__ msg) {
    long t = (long)blockIdx.x * blockDim.x + threadIdx.x;
    long edge = t >> 5;
    int  c    = (int)(t & 31) << 2;
    if (edge >= N_EDGES) return;
    long long d = dst_idx[edge];
    v4f v = __builtin_nontemporal_load((const v4f*)(e + edge * DIM + c));
    float* p = msg + (long)d * DIM + c;
    atomicAdd(p + 0, v[0]);
    atomicAdd(p + 1, v[1]);
    atomicAdd(p + 2, v[2]);
    atomicAdd(p + 3, v[3]);
}

// ---------------- WMMA operand loaders (layouts per CDNA5 ISA 7.12.2) --------
static __device__ inline v16bf load_a(const __bf16* act, int stride, int row0,
                                      int k0, int lane) {
    // A 16x32 bf16: lane<16 -> M=lane, K = k0+0..7 (v0-3) and k0+16..23 (v4-7)
    //               lane>=16 -> same M, K = k0+8..15 and k0+24..31
    int m  = lane & 15;
    int kb = k0 + ((lane & 16) ? 8 : 0);
    const __bf16* p = act + (row0 + m) * stride + kb;
    v8bf lo = *(const v8bf*)p;
    v8bf hi = *(const v8bf*)(p + 16);
    return __builtin_shufflevector(lo, hi, 0,1,2,3,4,5,6,7,8,9,10,11,12,13,14,15);
}

static __device__ inline v16bf load_b(const __bf16* wt, int stride, int n0,
                                      int k0, int lane) {
    // B 32x16 bf16: lane<16 -> N=lane, K=k0+0..7 and k0+16..23; lane>=16 -> N=lane-16, K=k0+8..15 and k0+24..31
    // weights stored transposed [out][in] so the per-lane K run is contiguous
    int n  = lane & 15;
    int kb = k0 + ((lane & 16) ? 8 : 0);
    const __bf16* p = wt + (n0 + n) * stride + kb;
    v8bf lo = *(const v8bf*)p;
    v8bf hi = *(const v8bf*)(p + 16);
    return __builtin_shufflevector(lo, hi, 0,1,2,3,4,5,6,7,8,9,10,11,12,13,14,15);
}

// ---------------- kernel 4: fused MLP + GroupNorm + residual ------------------
__global__ __launch_bounds__(256)
void fused_kernel(const float* __restrict__ x,   const float* __restrict__ msg,
                  const __bf16* __restrict__ wimg,
                  const float* __restrict__ b0,  const float* __restrict__ bh,
                  const float* __restrict__ bo,
                  const float* __restrict__ gnw, const float* __restrict__ gnb,
                  float* __restrict__ out) {
    extern __shared__ __align__(16) char smem[];
    __bf16* w0t  = (__bf16*)smem;                  // [128][S0]  (matches wimg order)
    __bf16* w1t  = w0t + 128 * S0;                 // [128][SW]
    __bf16* w2t  = w1t + 128 * SW;                 // [128][SW]
    __bf16* wot  = w2t + 128 * SW;                 // [128][SW]
    __bf16* act0 = wot + 128 * SW;                 // [ROWS][S0]
    __bf16* act1 = act0 + ROWS * S0;               // [ROWS][S0]
    float*  hbuf = (float*)(act1 + ROWS * S0);     // [ROWS][SH] fp32 staging

    const int tid  = threadIdx.x;
    const int lane = tid & 31;
    const int wave = tid >> 5;
    const int base = blockIdx.x * ROWS;

#if HAVE_TDM
    // --- TDM: DMA the 172032-B weight image (global, L2-resident) into LDS ---
    // D# per ISA 8.3/8.4: 1-D tile, data_size=8B, dim0 = 172032/8 = 21504.
    if (wave == 0) {
        unsigned lds_off = (unsigned)(uintptr_t)(void*)smem;  // addr[31:0] == LDS offset
        unsigned long long ga = (unsigned long long)(uintptr_t)wimg;
        const unsigned D0 = WIMG_BYTES / 8;                   // 21504
        u32x4 g0 = { 1u,                                      // count=1, user mode
                     lds_off,                                 // lds_addr
                     (unsigned)ga,                            // global_addr[31:0]
                     (unsigned)((ga >> 32) & 0x01FFFFFFu) | 0x80000000u }; // addr[56:32] | type=2
        i32x8 g1 = { (int)(3u << 16),                         // data_size=8B, mask=0
                     (int)((D0 & 0xFFFFu) << 16),             // tensor_dim0 lo16
                     (int)(((D0 >> 16) & 0xFFFFu) | (1u << 16)), // dim0 hi16 | tensor_dim1=1 lo
                     (int)((D0 & 0xFFFFu) << 16),             // tile_dim0
                     1,                                       // tile_dim1=1, tile_dim2=0
                     (int)D0,                                 // tensor_dim0_stride lo32
                     0, 0 };
        i32x4 z4 = {0, 0, 0, 0};
#if __clang_major__ >= 23
        i32x8 z8 = {0, 0, 0, 0, 0, 0, 0, 0};
        __builtin_amdgcn_tensor_load_to_lds(g0, g1, z4, z4, z8, 0);
#else
        __builtin_amdgcn_tensor_load_to_lds(g0, g1, z4, z4, 0);
#endif
        __builtin_amdgcn_s_wait_tensorcnt(0);
    }
#else
    // fallback: cooperative b128 copy of the prepared weight image
    {
        const uint4* src = (const uint4*)wimg;
        uint4* dst = (uint4*)smem;
        for (int i = tid; i < WIMG_BYTES / 16; i += 256) dst[i] = src[i];
    }
#endif

    // --- load activations: [x || msg] -> act0, bf16, float4-vectorized ---
    for (int idx = tid; idx < ROWS * 64; idx += 256) {
        int r = idx >> 6, c4 = (idx & 63) << 2;
        int node = base + r;
        v4f v = {0.f, 0.f, 0.f, 0.f};
        if (node < N_NODES) {
            if (c4 < DIM) v = *(const v4f*)(x + (long)node * DIM + c4);
            else v = __builtin_nontemporal_load(
                         (const v4f*)(msg + (long)node * DIM + (c4 - DIM)));
        }
        v4bf o = { tobf(v[0]), tobf(v[1]), tobf(v[2]), tobf(v[3]) };
        *(v4bf*)(act0 + r * S0 + c4) = o;
    }
    __syncthreads();

    const int n0 = wave * 16;        // each wave owns one 16-wide column tile
    const int nn = lane & 15;
    const int mb0 = (lane & 16) ? 8 : 0;

    // ---- layer 0: 256 -> 128, ReLU ----
    for (int rt = 0; rt < 2; ++rt) {
        v8f acc = {};
        for (int k0 = 0; k0 < 256; k0 += 32) {
            v16bf a = load_a(act0, S0, rt * 16, k0, lane);
            v16bf b = load_b(w0t, S0, n0, k0, lane);
            acc = __builtin_amdgcn_wmma_f32_16x16x32_bf16(
                false, a, false, b, (short)0, acc, false, false);
        }
        float bias = b0[n0 + nn];
        int mbase = rt * 16 + mb0;
        for (int j = 0; j < 8; ++j) {
            float v = acc[j] + bias;
            v = v > 0.f ? v : 0.f;
            act1[(mbase + j) * S0 + n0 + nn] = tobf(v);
        }
    }
    __syncthreads();

    // ---- hidden layer 1: 128 -> 128, ReLU (act1 -> act0) ----
    for (int rt = 0; rt < 2; ++rt) {
        v8f acc = {};
        for (int k0 = 0; k0 < 128; k0 += 32) {
            v16bf a = load_a(act1, S0, rt * 16, k0, lane);
            v16bf b = load_b(w1t, SW, n0, k0, lane);
            acc = __builtin_amdgcn_wmma_f32_16x16x32_bf16(
                false, a, false, b, (short)0, acc, false, false);
        }
        float bias = bh[n0 + nn];
        int mbase = rt * 16 + mb0;
        for (int j = 0; j < 8; ++j) {
            float v = acc[j] + bias;
            v = v > 0.f ? v : 0.f;
            act0[(mbase + j) * S0 + n0 + nn] = tobf(v);
        }
    }
    __syncthreads();

    // ---- hidden layer 2: 128 -> 128, ReLU (act0 -> act1) ----
    for (int rt = 0; rt < 2; ++rt) {
        v8f acc = {};
        for (int k0 = 0; k0 < 128; k0 += 32) {
            v16bf a = load_a(act0, S0, rt * 16, k0, lane);
            v16bf b = load_b(w2t, SW, n0, k0, lane);
            acc = __builtin_amdgcn_wmma_f32_16x16x32_bf16(
                false, a, false, b, (short)0, acc, false, false);
        }
        float bias = bh[DIM + n0 + nn];
        int mbase = rt * 16 + mb0;
        for (int j = 0; j < 8; ++j) {
            float v = acc[j] + bias;
            v = v > 0.f ? v : 0.f;
            act1[(mbase + j) * S0 + n0 + nn] = tobf(v);
        }
    }
    __syncthreads();

    // ---- output layer: 128 -> 128, no ReLU, fp32 to hbuf ----
    for (int rt = 0; rt < 2; ++rt) {
        v8f acc = {};
        for (int k0 = 0; k0 < 128; k0 += 32) {
            v16bf a = load_a(act1, S0, rt * 16, k0, lane);
            v16bf b = load_b(wot, SW, n0, k0, lane);
            acc = __builtin_amdgcn_wmma_f32_16x16x32_bf16(
                false, a, false, b, (short)0, acc, false, false);
        }
        float bias = bo[n0 + nn];
        int mbase = rt * 16 + mb0;
        for (int j = 0; j < 8; ++j)
            hbuf[(mbase + j) * SH + n0 + nn] = acc[j] + bias;
    }
    __syncthreads();

    // ---- GroupNorm(1,128) per row + residual; wave handles 4 rows ----
    for (int rr = 0; rr < 4; ++rr) {
        int r = wave * 4 + rr;
        float v0 = hbuf[r * SH + lane];
        float v1 = hbuf[r * SH + lane + 32];
        float v2 = hbuf[r * SH + lane + 64];
        float v3 = hbuf[r * SH + lane + 96];
        float s = v0 + v1 + v2 + v3;
        float q = v0 * v0 + v1 * v1 + v2 * v2 + v3 * v3;
        for (int off = 16; off; off >>= 1) {
            s += __shfl_xor(s, off, 32);
            q += __shfl_xor(q, off, 32);
        }
        float mu   = s * (1.f / 128.f);
        float var  = q * (1.f / 128.f) - mu * mu;
        float rstd = rsqrtf(var + EPS);
        int node = base + r;
        if (node < N_NODES) {
            long o = (long)node * DIM;
            __builtin_nontemporal_store(
                x[o + lane]      + (v0 - mu) * rstd * gnw[lane]      + gnb[lane],      out + o + lane);
            __builtin_nontemporal_store(
                x[o + lane + 32] + (v1 - mu) * rstd * gnw[lane + 32] + gnb[lane + 32], out + o + lane + 32);
            __builtin_nontemporal_store(
                x[o + lane + 64] + (v2 - mu) * rstd * gnw[lane + 64] + gnb[lane + 64], out + o + lane + 64);
            __builtin_nontemporal_store(
                x[o + lane + 96] + (v3 - mu) * rstd * gnw[lane + 96] + gnb[lane + 96], out + o + lane + 96);
        }
    }
}

extern "C" void kernel_launch(void* const* d_in, const int* in_sizes, int n_in,
                              void* d_out, int out_size, void* d_ws, size_t ws_size,
                              hipStream_t stream) {
    const float*     x   = (const float*)d_in[0];
    const float*     e   = (const float*)d_in[1];
    const long long* ei  = (const long long*)d_in[2];   // int64 [2, E]
    const float*     W0  = (const float*)d_in[3];
    const float*     b0  = (const float*)d_in[4];
    const float*     Wh  = (const float*)d_in[5];
    const float*     bh  = (const float*)d_in[6];
    const float*     Wo  = (const float*)d_in[7];
    const float*     bo  = (const float*)d_in[8];
    const float*     gnw = (const float*)d_in[9];
    const float*     gnb = (const float*)d_in[10];
    float*  out  = (float*)d_out;
    float*  msg  = (float*)d_ws;                        // N_NODES*DIM f32 = 51.2 MB
    __bf16* wimg = (__bf16*)((char*)d_ws + MSG_BYTES);  // +172 KB weight image

    // 1) zero msg accumulator
    long msg_n = (long)N_NODES * DIM;
    int zb = (int)((msg_n / 4 + 255) / 256);
    zero_kernel<<<zb, 256, 0, stream>>>(msg, msg_n);

    // 2) one-shot weight transpose/convert into d_ws (L2-resident image)
    prep_weights<<<320, 256, 0, stream>>>(W0, Wh, Wo, wimg);

    // 3) scatter-sum edge features (dst row = second row of edge_index)
    long st = (long)N_EDGES * 32;
    int sb = (int)((st + 255) / 256);
    scatter_kernel<<<sb, 256, 0, stream>>>(e, ei + N_EDGES, msg);

    // 4) fused MLP + GroupNorm + residual
    size_t lds = (size_t)WIMG_BYTES + (size_t)(2 * ROWS * S0) * 2
               + (size_t)ROWS * SH * 4;                 // 222,720 bytes
    int fb = (N_NODES + ROWS - 1) / ROWS;               // 3125
    fused_kernel<<<fb, 256, lds, stream>>>(x, msg, wimg, b0, bh, bo,
                                           gnw, gnb, out);
}
